// SynthesisLayer_67534065762688
// MI455X (gfx1250) — matrix operand order repaired and checked by reading, not compiled, not run on hardware
//
#include <hip/hip_runtime.h>
#include <hip/hip_bf16.h>

// ---- problem constants ----
#define BB    8
#define CIN   512
#define COUT  512
#define HH    64
#define WW    64
#define HO    66              // conv output spatial (64 + 2*2 - 3 + 1)
#define MTOT  (HO*HO)         // 4356 pixels per (b) GEMM
#define HU    138             // after 2x upfirdn: (66*2-1) + 9 + 9 - 12 + 1
#define TAPS  12

typedef __attribute__((ext_vector_type(16))) _Float16 v16h;
typedef __attribute__((ext_vector_type(8)))  float    v8f;
typedef __attribute__((ext_vector_type(4)))  unsigned int u32x4;

union Frag16 { v16h h; u32x4 q[2]; };

// ---- CDNA5 async global->LDS path (probe via __has_builtin; sync fallback) ----
#if defined(__has_builtin)
#if __has_builtin(__builtin_amdgcn_global_load_async_to_lds_b128)
#define HAVE_ASYNC 1
#endif
#if __has_builtin(__builtin_amdgcn_s_wait_asynccnt)
#define HAVE_AWAIT 1
#endif
#endif

// exact parameter types per hipcc diagnostic: int4 (gcc vector_size spelling),
// arg0 in the global ("__device__") address space, arg1 LDS.
typedef int i32x4 __attribute__((vector_size(16)));
typedef __attribute__((address_space(1))) i32x4* async_gptr;
typedef __attribute__((address_space(3))) i32x4* async_lptr;

template <int N>
__device__ __forceinline__ void wait_async()
{
#if defined(HAVE_ASYNC) && defined(HAVE_AWAIT)
    __builtin_amdgcn_s_wait_asynccnt(N);
#elif defined(HAVE_ASYNC)
    if (N == 0) asm volatile("s_wait_asynccnt 0x0" ::: "memory");
    else        asm volatile("s_wait_asynccnt 0x4" ::: "memory");
#endif
}

// copy 64 bytes (32 halves) global -> LDS for this thread
__device__ __forceinline__ void stage_copy(const _Float16* __restrict__ src,
                                           _Float16* __restrict__ dst)
{
#if defined(HAVE_ASYNC)
    #pragma unroll
    for (int j = 0; j < 4; ++j)
        __builtin_amdgcn_global_load_async_to_lds_b128(
            (async_gptr)(src + j*8), (async_lptr)(dst + j*8), 0, 0);
#else
    u32x4 r[4];
    #pragma unroll
    for (int j = 0; j < 4; ++j) r[j] = *(const u32x4*)(src + j*8);
    #pragma unroll
    for (int j = 0; j < 4; ++j) *(u32x4*)(dst + j*8) = r[j];
#endif
}

// ---------------------------------------------------------------------------
// K1: styles[b][i] = (w_lat[b] . aff_w[i]) / sqrt(512) + aff_b[i], then
//     normalize by rsqrt(mean(styles^2)) over the whole (B,CIN) array.
// ---------------------------------------------------------------------------
__global__ __launch_bounds__(256)
void styles_kernel(const float* __restrict__ w_lat, const float* __restrict__ aff_w,
                   const float* __restrict__ aff_b, float* __restrict__ s_mod)
{
    __shared__ float st[BB*CIN];
    __shared__ float red[256];
    float lsum = 0.f;
    for (int idx = threadIdx.x; idx < BB*CIN; idx += 256) {
        int b = idx >> 9, i = idx & 511;
        const float* wl = w_lat + b*512;
        const float* aw = aff_w + (size_t)i*512;
        float acc = 0.f;
        for (int d = 0; d < 512; ++d) acc += wl[d] * aw[d];
        float v = acc * 0.044194173824159216f + aff_b[i];   // 1/sqrt(512)
        st[idx] = v;
        lsum += v*v;
    }
    red[threadIdx.x] = lsum; __syncthreads();
    for (int s = 128; s > 0; s >>= 1) {
        if (threadIdx.x < s) red[threadIdx.x] += red[threadIdx.x+s];
        __syncthreads();
    }
    float rs = rsqrtf(red[0] / (float)(BB*CIN));
    for (int idx = threadIdx.x; idx < BB*CIN; idx += 256) s_mod[idx] = st[idx]*rs;
}

// ---------------------------------------------------------------------------
// K2: per-out-channel weight stats + f16 weight pack to [kk][o][i] layout.
// ---------------------------------------------------------------------------
__global__ __launch_bounds__(256)
void wprep_kernel(const float* __restrict__ conv_w, float* __restrict__ wn,
                  float* __restrict__ wsq, _Float16* __restrict__ wt)
{
    int o = blockIdx.x;
    __shared__ float red[256];
    float lsum = 0.f;
    for (int i = threadIdx.x; i < CIN; i += 256) {
        const float* wp = conv_w + ((size_t)o*CIN + i)*9;
        float ss = 0.f;
        #pragma unroll
        for (int kk = 0; kk < 9; ++kk) {
            float v = wp[kk];
            ss += v*v;
            wt[((size_t)kk*COUT + o)*CIN + i] = (_Float16)v;
        }
        wsq[(size_t)o*CIN + i] = ss;
        lsum += ss;
    }
    red[threadIdx.x] = lsum; __syncthreads();
    for (int s = 128; s > 0; s >>= 1) {
        if (threadIdx.x < s) red[threadIdx.x] += red[threadIdx.x+s];
        __syncthreads();
    }
    if (threadIdx.x == 0) wn[o] = rsqrtf(red[0] / (float)(CIN*9));
}

// ---------------------------------------------------------------------------
// K2b: coef[b][o] = wn[o] * rsqrt(wn[o]^2 * sum_i s^2[b][i]*wsq[o][i] + 1e-8)
// ---------------------------------------------------------------------------
__global__ __launch_bounds__(256)
void coef_kernel(const float* __restrict__ s_mod, const float* __restrict__ wn,
                 const float* __restrict__ wsq, float* __restrict__ coef)
{
    int idx = blockIdx.x*256 + threadIdx.x;     // 4096 = B*COUT
    int b = idx >> 9, o = idx & 511;
    const float* sp = s_mod + b*CIN;
    const float* wp = wsq + (size_t)o*CIN;
    float acc = 0.f;
    for (int i = 0; i < CIN; ++i) { float s = sp[i]; acc += s*s*wp[i]; }
    float w = wn[o];
    coef[idx] = w * rsqrtf(w*w*acc + 1e-8f);
}

// ---------------------------------------------------------------------------
// K2c: modulate + NCHW->NHWC transpose + f16 pack: xm[b][y][x][i] = f16(x*s)
// ---------------------------------------------------------------------------
__global__ __launch_bounds__(256)
void modpack_kernel(const float* __restrict__ x, const float* __restrict__ s,
                    _Float16* __restrict__ xm)
{
    int b  = blockIdx.z;
    int cb = blockIdx.y * 32;
    int pb = blockIdx.x * 32;
    __shared__ float tile[32][33];
    int tx = threadIdx.x & 31;
    int ty = threadIdx.x >> 5;
    #pragma unroll
    for (int cc = 0; cc < 32; cc += 8) {
        int c = cb + ty + cc;
        tile[ty+cc][tx] = x[((size_t)(b*CIN + c))*(HH*WW) + pb + tx] * s[b*CIN + c];
    }
    __syncthreads();
    #pragma unroll
    for (int pp = 0; pp < 32; pp += 8) {
        int p = pb + ty + pp;
        xm[((size_t)b*(HH*WW) + p)*CIN + cb + tx] = (_Float16)tile[tx][ty+pp];
    }
}

// ---------------------------------------------------------------------------
// K3: implicit-GEMM modulated conv via WMMA f16->f32.
//   Block: 256 thr = 8 waves; tile M=64 pixels x N=128 outs.
//   K loop: 72 stages of 64 channels (9 taps x 8 blocks), B staged async
//   into a double-buffered, padded LDS tile (16KB logical per stage).
//   Epilogue transposes through LDS for coalesced NCHW stores.
// ---------------------------------------------------------------------------
#define SB_STRIDE 72          // 64 halves + 8 pad (kills bank conflicts)
#define NSTAGE    (9 * (CIN/64))   // 72

__global__ __launch_bounds__(256)
void conv_wmma_kernel(const _Float16* __restrict__ xm,   // [B][64][64][512] f16
                      const _Float16* __restrict__ wt,   // [9][512][512]   f16
                      const float* __restrict__ coef,    // [B][512]
                      const float* __restrict__ bias,    // [512]
                      float* __restrict__ y1)            // [B][512][66][66] f32
{
    const int b    = blockIdx.z;
    const int tm   = blockIdx.x;           // M tile (64 pixels)
    const int tn   = blockIdx.y;           // N tile (128 outs)
    const int tid  = threadIdx.x;
    const int wave = tid >> 5;
    const int lane = tid & 31;
    const int mw   = wave & 3;
    const int nw   = wave >> 2;
    const int half = lane >> 4;
    const int l16  = lane & 15;

    // A-side pixel for this lane (A layout: lane -> M row)
    const int p  = tm*64 + mw*16 + l16;
    const int yy = p / HO, xx = p % HO;

    __shared__ union {
        _Float16 sb[2][128*SB_STRIDE];     // B stage double buffer (36 KB)
        float    sc[64*129];               // epilogue transpose     (33 KB)
    } sh;

    // this thread's slice of a B stage: o = tid/2, 32 halves at ib
    const int st_o  = tid >> 1;
    const int st_ib = (tid & 1) * 32;
    const _Float16* wtn = wt + (size_t)tn*128*CIN;   // N-tile base

    v8f acc[4] = {};

    // prologue: issue stage 0
    stage_copy(wtn + (size_t)st_o*CIN + st_ib,
               &sh.sb[0][st_o*SB_STRIDE + st_ib]);

    for (int s = 0; s < NSTAGE; ++s) {
        const int kk = s >> 3;
        const int cb = (s & 7) << 6;

        if (s + 1 < NSTAGE) {
            const int kk2 = (s+1) >> 3;
            const int cb2 = ((s+1) & 7) << 6;
            stage_copy(wtn + ((size_t)kk2*COUT + st_o)*CIN + cb2 + st_ib,
                       &sh.sb[(s+1)&1][st_o*SB_STRIDE + st_ib]);
            wait_async<4>();               // stage s landed; s+1 in flight
        } else {
            wait_async<0>();               // drain
        }
        __syncthreads();                   // publish stage s to all waves

        const int iy = yy + (kk/3) - 2;
        const int ix = xx + (kk%3) - 2;
        const bool valid = (p < MTOT) && (iy >= 0) && (iy < HH) && (ix >= 0) && (ix < WW);
        const _Float16* abase = xm + (((size_t)(b*HH + iy))*WW + ix)*CIN + cb;
        const _Float16* sbb = sh.sb[s & 1];

        #pragma unroll
        for (int ks = 0; ks < 2; ++ks) {
            // A fragment: lane<16 holds K {0..7,16..23}, lane>=16 {8..15,24..31}
            Frag16 a;
            if (valid) {
                const _Float16* ap = abase + ks*32 + half*8;
                a.q[0] = *(const u32x4*)(ap);
                a.q[1] = *(const u32x4*)(ap + 16);
            } else {
                a.q[0] = (u32x4){0,0,0,0};
                a.q[1] = (u32x4){0,0,0,0};
            }
            #pragma unroll
            for (int s4 = 0; s4 < 4; ++s4) {
                // B fragment: lane -> N col (l16), lane<16 holds K {0..7,16..23}, lane>=16 {8..15,24..31}
                Frag16 bf;
                const _Float16* bp = sbb + (nw*64 + s4*16 + l16)*SB_STRIDE
                                         + ks*32 + half*8;
                bf.q[0] = *(const u32x4*)(bp);
                bf.q[1] = *(const u32x4*)(bp + 16);
                acc[s4] = __builtin_amdgcn_wmma_f32_16x16x32_f16(
                    false, a.h, false, bf.h, (short)0, acc[s4], false, false);
            }
        }
        __syncthreads();                   // readers done before buf reuse
    }

    // ---- epilogue: LDS transpose -> coalesced stores ----
    // D layout: lane -> N col (l16), vgpr r -> M row (half*8 + r)
    #pragma unroll
    for (int s4 = 0; s4 < 4; ++s4) {
        int o_local = nw*64 + s4*16 + l16;
        #pragma unroll
        for (int r = 0; r < 8; ++r) {
            int p_local = mw*16 + half*8 + r;
            sh.sc[p_local*129 + o_local] = acc[s4][r];
        }
    }
    __syncthreads();
    for (int idx = tid; idx < 64*128; idx += 256) {
        int o_local = idx >> 6;
        int p_local = idx & 63;
        int pe = tm*64 + p_local;
        if (pe < MTOT) {
            int o = tn*128 + o_local;
            float v = sh.sc[p_local*129 + o_local] * coef[b*COUT + o] + bias[o];
            v = fminf(fmaxf(v, -256.0f), 256.0f);
            y1[((size_t)(b*COUT + o))*MTOT + pe] = v;
        }
    }
}

// ---------------------------------------------------------------------------
// K4: fused upfirdn(up=2) -> lrelu*sqrt(2) -> clamp -> upfirdn(down=2),
//     one (b,c) 66x66 image per workgroup, intermediates in 130KB LDS.
// ---------------------------------------------------------------------------
__global__ __launch_bounds__(256)
void upfirdn_kernel(const float* __restrict__ y1, const float* __restrict__ up_f,
                    const float* __restrict__ dn_f, float* __restrict__ out)
{
    extern __shared__ float smem[];
    float* s_in = smem;                       // 66*66   = 4356
    float* s_t1 = smem + 4356;                // 66*138  = 9108
    float* s_t2 = smem + 4356 + 9108;         // 138*138 = 19044
    float* s_t3 = smem;                       // 138*64  = 8832 (reuses s_in/s_t1)

    const int img = blockIdx.x;               // b*512 + c
    const float* src = y1 + (size_t)img * MTOT;

    float fu[TAPS], fd[TAPS];
    #pragma unroll
    for (int t = 0; t < TAPS; ++t) {          // f[::-1] * sqrt(gain)
        fu[t] = up_f[TAPS-1-t] * 2.0f;        // up gain = 4 -> sqrt = 2 per pass
        fd[t] = dn_f[TAPS-1-t];
    }

    for (int i = threadIdx.x; i < MTOT; i += 256) s_in[i] = src[i];
    __syncthreads();

    // horizontal up: 66 x 138
    for (int i = threadIdx.x; i < HO*HU; i += 256) {
        int r = i / HU, c = i % HU;
        float acc = 0.f;
        #pragma unroll
        for (int t = 0; t < TAPS; ++t) {
            int j = c + t - 9;
            if ((j & 1) == 0) {
                int k = j >> 1;
                if (k >= 0 && k < HO) acc += fu[t] * s_in[r*HO + k];
            }
        }
        s_t1[i] = acc;
    }
    __syncthreads();

    // vertical up + lrelu + clamp: 138 x 138
    for (int i = threadIdx.x; i < HU*HU; i += 256) {
        int r = i / HU, c = i % HU;
        float acc = 0.f;
        #pragma unroll
        for (int t = 0; t < TAPS; ++t) {
            int j = r + t - 9;
            if ((j & 1) == 0) {
                int k = j >> 1;
                if (k >= 0 && k < HO) acc += fu[t] * s_t1[k*HU + c];
            }
        }
        acc = (acc < 0.f ? acc * 0.2f : acc) * 1.4142135623730951f;
        acc = fminf(fmaxf(acc, -256.0f), 256.0f);
        s_t2[i] = acc;
    }
    __syncthreads();

    // horizontal down (stride 2, no pad): 138 x 64
    for (int i = threadIdx.x; i < HU*64; i += 256) {
        int r = i >> 6, c = i & 63;
        float acc = 0.f;
        #pragma unroll
        for (int t = 0; t < TAPS; ++t) acc += fd[t] * s_t2[r*HU + 2*c + t];
        s_t3[i] = acc;
    }
    __syncthreads();

    // vertical down + store: 64 x 64
    float* dst = out + (size_t)img * (HH*WW);
    for (int i = threadIdx.x; i < HH*WW; i += 256) {
        int r = i >> 6, c = i & 63;
        float acc = 0.f;
        #pragma unroll
        for (int t = 0; t < TAPS; ++t) acc += fd[t] * s_t3[(2*r + t)*64 + c];
        dst[i] = acc;
    }
}

// ---------------------------------------------------------------------------
extern "C" void kernel_launch(void* const* d_in, const int* in_sizes, int n_in,
                              void* d_out, int out_size, void* d_ws, size_t ws_size,
                              hipStream_t stream)
{
    const float* x      = (const float*)d_in[0];
    const float* w_lat  = (const float*)d_in[1];
    const float* conv_w = (const float*)d_in[2];
    const float* conv_b = (const float*)d_in[3];
    const float* aff_w  = (const float*)d_in[4];
    const float* aff_b  = (const float*)d_in[5];
    const float* up_f   = (const float*)d_in[6];
    const float* dn_f   = (const float*)d_in[7];
    float* out = (float*)d_out;

    char* ws = (char*)d_ws;
    float*    s_mod = (float*)ws;     ws += (size_t)BB*CIN*4;          // 16 KB
    float*    wn    = (float*)ws;     ws += (size_t)COUT*4;            // 2 KB
    float*    wsq   = (float*)ws;     ws += (size_t)COUT*CIN*4;        // 1 MB
    float*    coef  = (float*)ws;     ws += (size_t)BB*COUT*4;         // 16 KB
    _Float16* wt    = (_Float16*)ws;  ws += (size_t)9*COUT*CIN*2;      // 4.7 MB
    _Float16* xm    = (_Float16*)ws;  ws += (size_t)BB*HH*WW*CIN*2;    // 33.6 MB
    float*    y1    = (float*)ws;     ws += (size_t)BB*COUT*MTOT*4;    // 71.4 MB

    styles_kernel<<<1, 256, 0, stream>>>(w_lat, aff_w, aff_b, s_mod);
    wprep_kernel<<<COUT, 256, 0, stream>>>(conv_w, wn, wsq, wt);
    coef_kernel<<<(BB*COUT)/256, 256, 0, stream>>>(s_mod, wn, wsq, coef);

    dim3 gmod((HH*WW)/32, CIN/32, BB);
    modpack_kernel<<<gmod, 256, 0, stream>>>(x, s_mod, xm);

    dim3 gconv((MTOT + 63)/64, COUT/128, BB);                          // 69 x 4 x 8
    conv_wmma_kernel<<<gconv, 256, 0, stream>>>(xm, wt, coef, conv_b, y1);

    size_t lds_bytes = (size_t)(4356 + 9108 + 19044) * sizeof(float);  // ~130 KB
    upfirdn_kernel<<<BB*COUT, 256, lds_bytes, stream>>>(y1, up_f, dn_f, out);
}
